// RankedListLoss_10256381903181
// MI455X (gfx1250) — matrix operand order repaired and checked by reading, not compiled, not run on hardware
//
#include <hip/hip_runtime.h>

typedef __attribute__((ext_vector_type(16))) __bf16 v16bf;
typedef __attribute__((ext_vector_type(8)))  float  v8f;

#define N_ROWS 4096
#define DIMK   512
#define MARGIN 0.4f
#define ALPHA  1.2f
#define TVAL   10.0f
#define LOG2E  1.44269504088896340736f

static __device__ __forceinline__ unsigned short f32_to_bf16_rne(float f) {
    unsigned u = __builtin_bit_cast(unsigned, f);
    unsigned r = u + 0x7FFFu + ((u >> 16) & 1u);
    return (unsigned short)(r >> 16);
}
static __device__ __forceinline__ float bf16_to_f32(unsigned short h) {
    unsigned u = ((unsigned)h) << 16;
    return __builtin_bit_cast(float, u);
}

// Branchless ranked-list accumulation for one distance element.
// Keeps EXEC fully populated (no saveexec regions inside the hot loop).
static __device__ __forceinline__ void rll_accum(float c, float sqi_r, float sqj,
                                                 int li_r, int lj, int irow_r, int jb,
                                                 float& ap, float& cnt,
                                                 float& an, float& wsum) {
    const float d2   = sqi_r + sqj - 2.0f * c;
    // d2 clamped >= 1e-12 (no denormals) -> raw v_sqrt_f32 is accurate here.
    const float dist = __builtin_amdgcn_sqrtf(fmaxf(d2, 1e-12f));
    const bool  bs   = (li_r == lj);
    const float mpos = (bs && (irow_r != jb)) ? 1.0f : 0.0f;  // positive, off-diag
    const float mneg = bs ? 0.0f : 1.0f;                      // negative pair
    ap  = fmaf(mpos, fmaxf(dist - (ALPHA - MARGIN), 0.0f), ap);
    cnt += mpos;
    const float t = ALPHA - dist;
    const float e = __builtin_amdgcn_exp2f((TVAL * LOG2E) * t);
    const float w = (t > 0.0f) ? (mneg * e) : 0.0f;
    an   = fmaf(w, t, an);
    wsum += w;
}

// ---------------------------------------------------------------------------
// Kernel 1: fp32 -> bf16 hi/lo split + row squared norms
// ---------------------------------------------------------------------------
__global__ void __launch_bounds__(256)
rll_prep_kernel(const float* __restrict__ feats,
                unsigned short* __restrict__ hiW,
                unsigned short* __restrict__ loW,
                float* __restrict__ sqW) {
    __shared__ float red[256];
    const int row = blockIdx.x;
    const int tid = threadIdx.x;
    float acc = 0.0f;
#pragma unroll
    for (int e = 0; e < 2; ++e) {
        const int k = tid + e * 256;
        const float f = feats[(size_t)row * DIMK + k];
        const unsigned short h = f32_to_bf16_rne(f);
        const float fh = bf16_to_f32(h);
        const unsigned short l = f32_to_bf16_rne(f - fh);
        hiW[(size_t)row * DIMK + k] = h;
        loW[(size_t)row * DIMK + k] = l;
        acc += f * f;
    }
    red[tid] = acc;
    __syncthreads();
    for (int s = 128; s > 0; s >>= 1) {
        if (tid < s) red[tid] += red[tid + s];
        __syncthreads();
    }
    if (tid == 0) sqW[row] = red[0];
}

// ---------------------------------------------------------------------------
// Kernel 2: fused Gram (bf16x3 WMMA) + distance + ranked-list row reductions
// Workgroup = 256 threads = 8 waves laid out as 2 (M) x 4 (N) groups.
// Each wave computes a 16(M) x 32(N) stripe (two 16x16 accumulators) so one
// A fragment feeds two B tiles: 12 b128 loads per 6 WMMAs.
// Workgroup owns 32 rows, sweeps all 4096 columns in 128-wide chunks.
// ---------------------------------------------------------------------------
__global__ void __launch_bounds__(256, 1)
rll_main_kernel(const unsigned short* __restrict__ hiW,
                const unsigned short* __restrict__ loW,
                const float* __restrict__ sqW,
                const int* __restrict__ labels,
                float* __restrict__ apW,
                float* __restrict__ cntW,
                float* __restrict__ anW,
                float* __restrict__ wW) {
    __shared__ float s_ap[4][32], s_cnt[4][32], s_an[4][32], s_w[4][32];

    const int i0    = blockIdx.x * 32;
    const int wave  = threadIdx.x >> 5;
    const int lane  = threadIdx.x & 31;
    const int mt    = wave >> 2;     // 0..1 : which 16-row tile
    const int nt    = wave & 3;      // 0..3 : which 32-col stripe in the 128 chunk
    const int lhalf = lane >> 4;     // 0/1  : lane half (K-half for A/B frags)
    const int l15   = lane & 15;

    // C layout: VGPR r, lane L -> row = r + 8*lhalf, col = l15
    int   irow[8];
    int   li[8];
    float sqi[8];
#pragma unroll
    for (int r = 0; r < 8; ++r) {
        const int i = i0 + mt * 16 + r + 8 * lhalf;
        irow[r] = i;
        li[r]   = labels[i];
        sqi[r]  = sqW[i];
    }

    float rs_ap[8], rs_cnt[8], rs_an[8], rs_w[8];
#pragma unroll
    for (int r = 0; r < 8; ++r) { rs_ap[r] = 0.f; rs_cnt[r] = 0.f; rs_an[r] = 0.f; rs_w[r] = 0.f; }

    const int ia = i0 + mt * 16 + l15;   // A-fragment row for this lane

    for (int j0 = 0; j0 < N_ROWS; j0 += 128) {
        const int jb0 = j0 + nt * 32 + l15;       // B row for tile 0
        const int jb1 = jb0 + 16;                 // B row for tile 1
        const int   lj0  = labels[jb0];
        const int   lj1  = labels[jb1];
        const float sqj0 = sqW[jb0];
        const float sqj1 = sqW[jb1];

        v8f acc0 = {};
        v8f acc1 = {};

#pragma unroll 2
        for (int kc = 0; kc < DIMK; kc += 32) {
            // A 16x32 bf16 frag: lane holds row ia; regs v0..7 -> k=kc+16*lhalf..+15
            // => 32 contiguous bytes = two 16B loads.
            const size_t aoff = (size_t)ia * DIMK + kc + lhalf * 16;
            // B 32x16 bf16 frag: lane holds row jb; regs v0..7 -> k=kc+16*lhalf..+15
            // => 32 contiguous bytes = two 16B loads.
            const size_t boff0 = (size_t)jb0 * DIMK + kc + lhalf * 16;
            const size_t boff1 = (size_t)jb1 * DIMK + kc + lhalf * 16;

            union { v16bf v; uint4 q[2]; } ah, al, bh0, bl0, bh1, bl1;
            ah.q[0]  = *(const uint4*)(hiW + aoff);
            ah.q[1]  = *(const uint4*)(hiW + aoff + 8);
            al.q[0]  = *(const uint4*)(loW + aoff);
            al.q[1]  = *(const uint4*)(loW + aoff + 8);
            bh0.q[0] = *(const uint4*)(hiW + boff0);
            bh0.q[1] = *(const uint4*)(hiW + boff0 + 8);
            bl0.q[0] = *(const uint4*)(loW + boff0);
            bl0.q[1] = *(const uint4*)(loW + boff0 + 8);
            bh1.q[0] = *(const uint4*)(hiW + boff1);
            bh1.q[1] = *(const uint4*)(hiW + boff1 + 8);
            bl1.q[0] = *(const uint4*)(loW + boff1);
            bl1.q[1] = *(const uint4*)(loW + boff1 + 8);

            // fp32-accurate product via bf16 split: hi*hi + hi*lo + lo*hi
            acc0 = __builtin_amdgcn_wmma_f32_16x16x32_bf16(false, ah.v, false, bh0.v,
                                                           (short)0, acc0, false, false);
            acc0 = __builtin_amdgcn_wmma_f32_16x16x32_bf16(false, ah.v, false, bl0.v,
                                                           (short)0, acc0, false, false);
            acc0 = __builtin_amdgcn_wmma_f32_16x16x32_bf16(false, al.v, false, bh0.v,
                                                           (short)0, acc0, false, false);
            acc1 = __builtin_amdgcn_wmma_f32_16x16x32_bf16(false, ah.v, false, bh1.v,
                                                           (short)0, acc1, false, false);
            acc1 = __builtin_amdgcn_wmma_f32_16x16x32_bf16(false, ah.v, false, bl1.v,
                                                           (short)0, acc1, false, false);
            acc1 = __builtin_amdgcn_wmma_f32_16x16x32_bf16(false, al.v, false, bh1.v,
                                                           (short)0, acc1, false, false);
        }

        // Fused, fully branchless epilogue (keeps EXEC all-ones).
#pragma unroll
        for (int r = 0; r < 8; ++r) {
            rll_accum(acc0[r], sqi[r], sqj0, li[r], lj0, irow[r], jb0,
                      rs_ap[r], rs_cnt[r], rs_an[r], rs_w[r]);
            rll_accum(acc1[r], sqi[r], sqj1, li[r], lj1, irow[r], jb1,
                      rs_ap[r], rs_cnt[r], rs_an[r], rs_w[r]);
        }
    }

    // Reduce across the 16 lanes of each half (columns of the 16x16 tiles).
#pragma unroll
    for (int r = 0; r < 8; ++r) {
#pragma unroll
        for (int off = 1; off < 16; off <<= 1) {
            rs_ap[r]  += __shfl_xor(rs_ap[r],  off, 32);
            rs_cnt[r] += __shfl_xor(rs_cnt[r], off, 32);
            rs_an[r]  += __shfl_xor(rs_an[r],  off, 32);
            rs_w[r]   += __shfl_xor(rs_w[r],   off, 32);
        }
    }
    if (l15 == 0) {
#pragma unroll
        for (int r = 0; r < 8; ++r) {
            const int rl = mt * 16 + r + 8 * lhalf;
            s_ap[nt][rl]  = rs_ap[r];
            s_cnt[nt][rl] = rs_cnt[r];
            s_an[nt][rl]  = rs_an[r];
            s_w[nt][rl]   = rs_w[r];
        }
    }
    __syncthreads();
    // Fixed-order cross-wave combine (deterministic, no FP atomics);
    // each row is owned by exactly one workgroup -> plain stores.
    if (threadIdx.x < 32) {
        const int rl = threadIdx.x;
        float a = 0.f, c = 0.f, n = 0.f, w = 0.f;
        for (int t = 0; t < 4; ++t) {
            a += s_ap[t][rl];
            c += s_cnt[t][rl];
            n += s_an[t][rl];
            w += s_w[t][rl];
        }
        apW[i0 + rl]  = a;
        cntW[i0 + rl] = c;
        anW[i0 + rl]  = n;
        wW[i0 + rl]   = w;
    }
}

// ---------------------------------------------------------------------------
// Kernel 3: per-row loss + mean
// ---------------------------------------------------------------------------
__global__ void __launch_bounds__(256)
rll_finalize_kernel(const float* __restrict__ apW,
                    const float* __restrict__ cntW,
                    const float* __restrict__ anW,
                    const float* __restrict__ wW,
                    float* __restrict__ out) {
    __shared__ float red[256];
    const int tid = threadIdx.x;
    float acc = 0.0f;
    for (int i = tid; i < N_ROWS; i += 256) {
        const float lap = apW[i] / (cntW[i] + 1e-5f);
        const float lan = anW[i] / (wW[i] + 1e-5f);
        acc += lap + lan;
    }
    red[tid] = acc;
    __syncthreads();
    for (int s = 128; s > 0; s >>= 1) {
        if (tid < s) red[tid] += red[tid + s];
        __syncthreads();
    }
    if (tid == 0) out[0] = red[0] / (float)N_ROWS;
}

// ---------------------------------------------------------------------------
extern "C" void kernel_launch(void* const* d_in, const int* in_sizes, int n_in,
                              void* d_out, int out_size, void* d_ws, size_t ws_size,
                              hipStream_t stream) {
    const float* feats  = (const float*)d_in[0];
    const int*   labels = (const int*)d_in[1];
    float*       out    = (float*)d_out;

    // Workspace layout: bf16 hi (4MB) | bf16 lo (4MB) | sq | ap | cnt | an | w
    unsigned short* hiW = (unsigned short*)d_ws;
    unsigned short* loW = hiW + (size_t)N_ROWS * DIMK;
    float* sqW  = (float*)(loW + (size_t)N_ROWS * DIMK);
    float* apW  = sqW  + N_ROWS;
    float* cntW = apW  + N_ROWS;
    float* anW  = cntW + N_ROWS;
    float* wW   = anW  + N_ROWS;

    rll_prep_kernel<<<N_ROWS, 256, 0, stream>>>(feats, hiW, loW, sqW);
    rll_main_kernel<<<N_ROWS / 32, 256, 0, stream>>>(hiW, loW, sqW, labels,
                                                     apW, cntW, anW, wW);
    rll_finalize_kernel<<<1, 256, 0, stream>>>(apW, cntW, anW, wW, out);
}
